// RecognitionLattice_81604378624798
// MI455X (gfx1250) — matrix operand order repaired and checked by reading, not compiled, not run on hardware
//
#include <hip/hip_runtime.h>
#include <math.h>

// ---------------- problem constants ----------------
#define Bc 8
#define Tc 128
#define Dc 512
#define Hc 512
#define Vc 64
#define Cc 65          // V+1
#define Uc 32
#define Cp 80          // C padded to 5*16 for WMMA tiling
#define NEG_INF_F (-1e30f)

typedef __attribute__((ext_vector_type(16))) __bf16 v16bf;
typedef __attribute__((ext_vector_type(8)))  float  v8f;

// ---------------- conversion kernels ----------------
__global__ void cvt_plain_bf16(const float* __restrict__ src, __bf16* __restrict__ dst, int n) {
  int i = blockIdx.x * blockDim.x + threadIdx.x;
  if (i < n) dst[i] = (__bf16)src[i];
}

// W_frame (D,H) f32 -> WfT (H,D) bf16   (transpose so B-fragment K is contiguous per lane)
__global__ void cvt_wfT(const float* __restrict__ wf, __bf16* __restrict__ wfT) {
  int i = blockIdx.x * blockDim.x + threadIdx.x;
  if (i < Dc * Hc) {
    int d = i / Hc, h = i % Hc;
    wfT[h * Dc + d] = (__bf16)wf[i];
  }
}

// context_embed (C,H) f32 -> (Cp,H) bf16, rows >= C zero-padded
__global__ void cvt_ce(const float* __restrict__ ce, __bf16* __restrict__ ceB) {
  int i = blockIdx.x * blockDim.x + threadIdx.x;
  if (i < Cp * Hc) {
    int c = i / Hc, h = i % Hc;
    ceB[i] = (c < Cc) ? (__bf16)ce[c * Hc + h] : (__bf16)0.0f;
  }
}

// W_out (H,C) f32 -> WoT (Cp,H) bf16, cols (rows of WoT) >= C zero-padded
__global__ void cvt_woT(const float* __restrict__ wo, __bf16* __restrict__ woT) {
  int i = blockIdx.x * blockDim.x + threadIdx.x;
  if (i < Cp * Hc) {
    int k = i / Hc, h = i % Hc;
    woT[i] = (k < Cc) ? (__bf16)wo[h * Cc + k] : (__bf16)0.0f;
  }
}

// ---------------- WMMA helpers ----------------
// A fragment 16x32 bf16 (M x K): lanes 0-15 -> rows 0..15 with K in {0..7,16..23};
// lanes 16-31 -> same rows with K in {8..15,24..31}. Element e: K = (e<8 ? e : e+8) + half*8.
// B fragment 32x16 bf16 (K x N): lane holds column n = lane%16, K = (e<8 ? e : e+8) + half*8 (same as A).
// D fragment 16x16 f32: element r -> row r + half*8, col lane%16.

__device__ __forceinline__ v8f wmma_bf16(v16bf a, v16bf b, v8f c) {
  return __builtin_amdgcn_wmma_f32_16x16x32_bf16(false, a, false, b, (short)0, c, false, false);
}

// Load a raw bf16 A fragment (two contiguous 8-element runs -> 2x b128 loads)
__device__ __forceinline__ v16bf loadA_raw(const __bf16* __restrict__ base) {
  v16bf a;
  #pragma unroll
  for (int e = 0; e < 8; ++e) a[e] = base[e];
  #pragma unroll
  for (int e = 0; e < 8; ++e) a[8 + e] = base[16 + e];
  return a;
}

// Load a bf16 B fragment (two contiguous 8-element runs -> 2x b128 loads)
__device__ __forceinline__ v16bf loadB_raw(const __bf16* __restrict__ base) {
  v16bf b;
  #pragma unroll
  for (int e = 0; e < 8; ++e) b[e] = base[e];
  #pragma unroll
  for (int e = 0; e < 8; ++e) b[8 + e] = base[16 + e];
  return b;
}

// ---------------- proj = frames @ W_frame ----------------
// grid.x = (B*T)/16 = 64 blocks, 256 threads = 8 waves; wave w owns N-tiles w, w+8, w+16, w+24.
// Software-pipelined: fragments for chunk kc+1 are loaded before the WMMAs of chunk kc issue.
__global__ __launch_bounds__(256) void proj_gemm(const __bf16* __restrict__ fB,
                                                 const __bf16* __restrict__ wfT,
                                                 float* __restrict__ proj) {
  const int tid  = threadIdx.x;
  const int wave = tid >> 5;
  const int lane = tid & 31;
  const int half = lane >> 4;
  const int lm   = lane & 15;
  const int m0   = blockIdx.x * 16;

  const __bf16* aBase = fB + (size_t)(m0 + lm) * Dc + half * 8;
  const __bf16* bBase[4];
  #pragma unroll
  for (int j = 0; j < 4; ++j)
    bBase[j] = wfT + (size_t)((wave + 8 * j) * 16 + lm) * Dc + half * 8;

  v8f acc[4] = {};
  v16bf a0, a1, b0[4], b1[4];

  a0 = loadA_raw(aBase);
  #pragma unroll
  for (int j = 0; j < 4; ++j) b0[j] = loadB_raw(bBase[j]);

  constexpr int KC = Dc / 32;   // 16 chunks
  #pragma unroll 2
  for (int kc = 0; kc < KC; kc += 2) {
    // prefetch chunk kc+1
    a1 = loadA_raw(aBase + (kc + 1) * 32);
    #pragma unroll
    for (int j = 0; j < 4; ++j) b1[j] = loadB_raw(bBase[j] + (kc + 1) * 32);
    #pragma unroll
    for (int j = 0; j < 4; ++j) acc[j] = wmma_bf16(a0, b0[j], acc[j]);

    // prefetch chunk kc+2 (wraps harmlessly on last iteration: stays in-bounds pad-free
    // because we clamp the index)
    const int kn = (kc + 2 < KC) ? (kc + 2) : kc;
    a0 = loadA_raw(aBase + kn * 32);
    #pragma unroll
    for (int j = 0; j < 4; ++j) b0[j] = loadB_raw(bBase[j] + kn * 32);
    #pragma unroll
    for (int j = 0; j < 4; ++j) acc[j] = wmma_bf16(a1, b1[j], acc[j]);
  }

  #pragma unroll
  for (int j = 0; j < 4; ++j) {
    const int n0 = (wave + 8 * j) * 16;
    #pragma unroll
    for (int r = 0; r < 8; ++r) {
      const int row = m0 + r + half * 8;
      proj[(size_t)row * Hc + n0 + lm] = acc[j][r];
    }
  }
}

// ---------------- logits[b,t] = tanh(proj + ce) @ W_out ----------------
// grid.x = B*T blocks; 160 threads = 5 waves; wave m owns M-tile m (rows 16m..16m+15 of joint),
// loops over 5 N-tiles. B fragments double-buffered so L2 loads overlap the tanh VALU work
// and the WMMAs of the previous chunk. Output stored as logits[t][b][c][k], c,k < 65.
__global__ __launch_bounds__(160) void logits_fused(const float* __restrict__ proj,
                                                    const __bf16* __restrict__ ceB,
                                                    const __bf16* __restrict__ woT,
                                                    float* __restrict__ logits) {
  const int bt = blockIdx.x;
  const int b  = bt / Tc;
  const int t  = bt % Tc;
  const int tid  = threadIdx.x;
  const int wave = tid >> 5;     // 0..4 == M tile
  const int lane = tid & 31;
  const int half = lane >> 4;
  const int lm   = lane & 15;

  __shared__ float projs[Hc];
  for (int i = tid; i < Hc; i += 160) projs[i] = proj[(size_t)bt * Hc + i];
  __syncthreads();

  const int m0 = wave * 16;
  const int mrow = m0 + lm;
  const __bf16* ceRow = ceB + (size_t)mrow * Hc + half * 8;
  const __bf16* bBase[5];
  #pragma unroll
  for (int n = 0; n < 5; ++n)
    bBase[n] = woT + (size_t)(n * 16 + lm) * Hc + half * 8;

  v8f acc[5] = {};
  v16bf b0[5], b1[5];

  #pragma unroll
  for (int n = 0; n < 5; ++n) b0[n] = loadB_raw(bBase[n]);

  constexpr int KC = Hc / 32;   // 16 chunks
  #pragma unroll 2
  for (int kc = 0; kc < KC; kc += 2) {
    // issue B loads for chunk kc+1; they fly while we do tanh + WMMA of chunk kc
    #pragma unroll
    for (int n = 0; n < 5; ++n) b1[n] = loadB_raw(bBase[n] + (kc + 1) * 32);

    {
      const int k0 = kc * 32;
      const int base = k0 + half * 8;
      v16bf a;
      #pragma unroll
      for (int e = 0; e < 8; ++e)
        a[e] = (__bf16)tanhf(projs[base + e] + (float)ceRow[k0 + e]);
      #pragma unroll
      for (int e = 0; e < 8; ++e)
        a[8 + e] = (__bf16)tanhf(projs[base + 16 + e] + (float)ceRow[k0 + 16 + e]);
      #pragma unroll
      for (int n = 0; n < 5; ++n) acc[n] = wmma_bf16(a, b0[n], acc[n]);
    }

    // issue B loads for chunk kc+2
    const int kn = (kc + 2 < KC) ? (kc + 2) : kc;
    #pragma unroll
    for (int n = 0; n < 5; ++n) b0[n] = loadB_raw(bBase[n] + kn * 32);

    {
      const int k0 = (kc + 1) * 32;
      const int base = k0 + half * 8;
      v16bf a;
      #pragma unroll
      for (int e = 0; e < 8; ++e)
        a[e] = (__bf16)tanhf(projs[base + e] + (float)ceRow[k0 + e]);
      #pragma unroll
      for (int e = 0; e < 8; ++e)
        a[8 + e] = (__bf16)tanhf(projs[base + 16 + e] + (float)ceRow[k0 + 16 + e]);
      #pragma unroll
      for (int n = 0; n < 5; ++n) acc[n] = wmma_bf16(a, b1[n], acc[n]);
    }
  }

  float* Lrow = logits + ((size_t)t * Bc + b) * (Cc * Cc);
  #pragma unroll
  for (int n = 0; n < 5; ++n) {
    #pragma unroll
    for (int r = 0; r < 8; ++r) {
      const int c = m0 + r + half * 8;
      const int k = n * 16 + lm;
      if (c < Cc && k < Cc) Lrow[c * Cc + k] = acc[n][r];
    }
  }
}

// ---------------- lattice scan (numerator + denominator) ----------------
__device__ __forceinline__ float lae(float a, float b) {
  float m = fmaxf(a, b);
  return m + logf(expf(a - m) + expf(b - m));
}

// grid = B blocks, 128 threads. Threads 0..64: denominator states; 95..127: numerator states.
__global__ __launch_bounds__(128) void lattice_scan(const float* __restrict__ logits,
                                                    const int* __restrict__ num_frames,
                                                    const int* __restrict__ labels,
                                                    const int* __restrict__ num_labels,
                                                    float* __restrict__ out) {
  const int b = blockIdx.x;
  const int tid = threadIdx.x;

  __shared__ float aden[Cc], nden[Cc];
  __shared__ float anum[Uc + 1], nnum[Uc + 1];
  __shared__ int   lab[Uc + 1];   // ctx: lab[0]=0, lab[u]=labels[u-1]

  if (tid < Cc) aden[tid] = (tid == 0) ? 0.0f : NEG_INF_F;
  if (tid <= Uc) anum[tid] = (tid == 0) ? 0.0f : NEG_INF_F;
  if (tid == 0) lab[0] = 0;
  if (tid < Uc) lab[tid + 1] = labels[b * Uc + tid];
  const int nf = num_frames[b];
  const int nl = num_labels[b];
  __syncthreads();

  for (int t = 0; t < Tc; ++t) {
    const float* L = logits + ((size_t)t * Bc + b) * (Cc * Cc);
    if (t < nf) {
      // prefetch next timestep's lattice slice (global_prefetch_b8)
      if (t + 1 < nf && tid < Cc)
        __builtin_prefetch(logits + ((size_t)(t + 1) * Bc + b) * (Cc * Cc) + tid * Cc, 0, 1);

      // denominator recursion
      if (tid == 0) {
        nden[0] = aden[0] + L[0];                 // stay-only for context 0
      } else if (tid < Cc) {
        const int j = tid;
        const float stay = aden[j] + L[j * Cc];   // blank at context j
        float mx = stay;
        for (int c = 0; c < Cc; ++c) {
          float v = aden[c] + L[c * Cc + j];      // lex transition c -> j (label j)
          mx = fmaxf(mx, v);
        }
        float s = expf(stay - mx);
        for (int c = 0; c < Cc; ++c)
          s += expf(aden[c] + L[c * Cc + j] - mx);
        nden[j] = mx + logf(s);
      }
      // numerator recursion
      if (tid >= 95) {
        const int u = tid - 95;                   // 0..32
        const float blv = anum[u] + L[lab[u] * Cc];
        const float sh = (u == 0) ? NEG_INF_F
                                  : anum[u - 1] + L[lab[u - 1] * Cc + lab[u]];
        nnum[u] = lae(blv, sh);
      }
    }
    __syncthreads();
    if (t < nf) {
      if (tid < Cc) aden[tid] = nden[tid];
      if (tid >= 95) anum[tid - 95] = nnum[tid - 95];
    }
    __syncthreads();
  }

  if (tid == 0) {
    float mx = NEG_INF_F;
    for (int c = 0; c < Cc; ++c) mx = fmaxf(mx, aden[c]);
    float s = 0.0f;
    for (int c = 0; c < Cc; ++c) s += expf(aden[c] - mx);
    const float den = mx + logf(s);
    out[b] = den - anum[nl];
  }
}

// ---------------- launch ----------------
extern "C" void kernel_launch(void* const* d_in, const int* in_sizes, int n_in,
                              void* d_out, int out_size, void* d_ws, size_t ws_size,
                              hipStream_t stream) {
  const float* frames = (const float*)d_in[0];   // (B,T,D)
  const float* Wf     = (const float*)d_in[1];   // (D,H)
  const float* ce     = (const float*)d_in[2];   // (C,H)
  const float* Wo     = (const float*)d_in[3];   // (H,C)
  const int* num_frames = (const int*)d_in[4];   // (B,)
  const int* labels     = (const int*)d_in[5];   // (B,U)
  const int* num_labels = (const int*)d_in[6];   // (B,)
  float* out = (float*)d_out;                    // (B,)

  // workspace layout (bytes, 256-aligned regions)
  char* ws = (char*)d_ws;
  size_t off = 0;
  __bf16* framesB = (__bf16*)(ws + off); off += (size_t)Bc * Tc * Dc * 2;        // 1 MB
  __bf16* wfT     = (__bf16*)(ws + off); off += (size_t)Hc * Dc * 2;             // 0.5 MB
  __bf16* ceB     = (__bf16*)(ws + off); off += (size_t)Cp * Hc * 2;             // 80 KB
  __bf16* woT     = (__bf16*)(ws + off); off += (size_t)Cp * Hc * 2;             // 80 KB
  off = (off + 255) & ~(size_t)255;
  float* proj     = (float*)(ws + off);  off += (size_t)Bc * Tc * Hc * 4;        // 2 MB
  float* logits   = (float*)(ws + off);  off += (size_t)Tc * Bc * Cc * Cc * 4;   // 17.3 MB
  (void)ws_size; (void)in_sizes; (void)n_in; (void)out_size;

  const int nF = Bc * Tc * Dc;
  cvt_plain_bf16<<<(nF + 255) / 256, 256, 0, stream>>>(frames, framesB, nF);
  cvt_wfT<<<(Dc * Hc + 255) / 256, 256, 0, stream>>>(Wf, wfT);
  cvt_ce<<<(Cp * Hc + 255) / 256, 256, 0, stream>>>(ce, ceB);
  cvt_woT<<<(Cp * Hc + 255) / 256, 256, 0, stream>>>(Wo, woT);

  proj_gemm<<<(Bc * Tc) / 16, 256, 0, stream>>>(framesB, wfT, proj);
  logits_fused<<<Bc * Tc, 160, 0, stream>>>(proj, ceB, woT, logits);
  lattice_scan<<<Bc, 128, 0, stream>>>(logits, num_frames, labels, num_labels, out);
}
